// SelfAttention_76854144795117
// MI455X (gfx1250) — matrix operand order repaired and checked by reading, not compiled, not run on hardware
//
#include <hip/hip_runtime.h>
#include <hip/hip_bf16.h>
#include <stdint.h>

typedef __bf16 bf16_t;
typedef __attribute__((ext_vector_type(16))) __bf16 v16bf;
typedef __attribute__((ext_vector_type(8)))  __bf16 v8bf;
typedef __attribute__((ext_vector_type(4)))  __bf16 v4bf;
typedef __attribute__((ext_vector_type(8)))  float  v8f;

#define WMMA_BF16(a, b, c) \
    __builtin_amdgcn_wmma_f32_16x16x32_bf16(false, (a), false, (b), (short)0, (c), false, false)

// ---------------------------------------------------------------------------
// CDNA5 async global->LDS copy (ASYNCcnt path, bypasses VGPRs). 16B per lane.
// ---------------------------------------------------------------------------
__device__ __forceinline__ void async_copy_b128(uint32_t lds_off, const void* gaddr) {
    asm volatile("global_load_async_to_lds_b128 %0, %1, off"
                 :: "v"(lds_off), "v"((unsigned long long)(uintptr_t)gaddr)
                 : "memory");
}
template <int N>
__device__ __forceinline__ void wait_async() {
    asm volatile("s_wait_asynccnt %0" :: "i"(N) : "memory");
}
__device__ __forceinline__ uint32_t lds_addr32(const void* p) {
    return (uint32_t)(uintptr_t)p;   // low 32 bits of generic LDS ptr = LDS offset
}

// ---------------------------------------------------------------------------
// WMMA fragment loaders (CDNA5 ISA §7.12.2, wave32)
// ---------------------------------------------------------------------------
__device__ __forceinline__ v16bf load_frag_a(const bf16_t* base, int stride, int lane) {
    const int r  = lane & 15;
    const int kb = (lane >> 4) << 3;
    const bf16_t* p = base + r * stride + kb;
    v8bf lo = *(const v8bf*)(p);
    v8bf hi = *(const v8bf*)(p + 16);
    v16bf a;
#pragma unroll
    for (int i = 0; i < 8; ++i) { a[i] = lo[i]; a[8 + i] = hi[i]; }
    return a;
}
__device__ __forceinline__ v16bf load_frag_b(const bf16_t* base, int stride, int lane) {
    const int n  = lane & 15;
    const int kb = (lane >> 4) << 3;
    const bf16_t* p = base + n * stride + kb;
    v8bf lo = *(const v8bf*)(p);
    v8bf hi = *(const v8bf*)(p + 16);
    v16bf b;
#pragma unroll
    for (int i = 0; i < 8; ++i) { b[i] = lo[i]; b[8 + i] = hi[i]; }
    return b;
}

// ---------------------------------------------------------------------------
// One-time converts (fp32 -> bf16; weights tile-transposed to [n][k])
// ---------------------------------------------------------------------------
__global__ __launch_bounds__(256) void cvt_bf16(const float* __restrict__ in,
                                                bf16_t* __restrict__ out, int n4) {
    const int i = blockIdx.x * 256 + threadIdx.x;
    if (i < n4) {
        const float4 f = reinterpret_cast<const float4*>(in)[i];
        v4bf o; o[0] = (bf16_t)f.x; o[1] = (bf16_t)f.y; o[2] = (bf16_t)f.z; o[3] = (bf16_t)f.w;
        *reinterpret_cast<v4bf*>(out + 4 * (size_t)i) = o;
    }
}

__global__ __launch_bounds__(256) void transpose_cvt(const float* __restrict__ in,
                                                     bf16_t* __restrict__ out,
                                                     int R, int C) {
    __shared__ float t[32][33];
    const int c  = blockIdx.x * 32 + threadIdx.x;
    const int r0 = blockIdx.y * 32;
#pragma unroll
    for (int j = threadIdx.y; j < 32; j += 8)
        t[j][threadIdx.x] = in[(size_t)(r0 + j) * C + c];
    __syncthreads();
    const int oc  = r0 + threadIdx.x;
    const int or0 = blockIdx.x * 32;
#pragma unroll
    for (int j = threadIdx.y; j < 32; j += 8)
        out[(size_t)(or0 + j) * R + oc] = (bf16_t)t[threadIdx.x][j];
}

// ---------------------------------------------------------------------------
// GEMM + bias: C[M,N] = A[M,K] @ Bt[N,K]^T + bias[N]. bf16 in, bf16/f32 out.
// Tiles 128x64x64, 256 threads = 8 waves, wave tile 32x32.
// Double-buffered async global->LDS pipeline: stage i+1 copies overlap
// stage i WMMAs (per-wave ASYNCcnt is in-order -> wait<=6 drains stage i).
// ---------------------------------------------------------------------------
#define GBM 128
#define GBN 64
#define GBK 64
#define ASTR (GBK + 8)          // 72 halves
#define BSTR (GBK + 8)
#define ASZ  (GBM * ASTR)
#define BSZ  (GBN * BSTR)

__device__ __forceinline__ void gemm_stage(const bf16_t* A, const bf16_t* Bt,
                                           int bm, int bn, int K, int k0,
                                           uint32_t as, uint32_t bs, int tid) {
    // A tile: 128 rows x 128B -> 1024 x 16B chunks, 4 per thread
#pragma unroll
    for (int i = 0; i < 4; ++i) {
        const int c = tid + i * 256;
        const int r = c >> 3, seg = c & 7;
        async_copy_b128(as + (uint32_t)(r * ASTR + seg * 8) * 2,
                        A + (size_t)(bm + r) * K + k0 + seg * 8);
    }
    // B tile: 64 rows x 128B -> 512 chunks, 2 per thread
#pragma unroll
    for (int i = 0; i < 2; ++i) {
        const int c = tid + i * 256;
        const int r = c >> 3, seg = c & 7;
        async_copy_b128(bs + (uint32_t)(r * BSTR + seg * 8) * 2,
                        Bt + (size_t)(bn + r) * K + k0 + seg * 8);
    }
}

template <bool OUT_BF16>
__global__ __launch_bounds__(256) void gemm_bias_wmma(const bf16_t* __restrict__ A,
                                                      const bf16_t* __restrict__ Bt,
                                                      const float* __restrict__ bias,
                                                      void* __restrict__ Cout,
                                                      int M, int N, int K) {
    __shared__ __align__(16) bf16_t As[2 * ASZ];
    __shared__ __align__(16) bf16_t Bs[2 * BSZ];   // [n][k]

    const int tid  = threadIdx.x;
    const int lane = tid & 31;
    const int wave = tid >> 5;
    const int wm   = wave >> 1;
    const int wn   = wave & 1;
    const int bm   = blockIdx.y * GBM;
    const int bn   = blockIdx.x * GBN;
    const uint32_t as0 = lds_addr32(As);
    const uint32_t bs0 = lds_addr32(Bs);

    v8f acc[2][2] = {};

    const int nst = K / GBK;
    gemm_stage(A, Bt, bm, bn, K, 0, as0, bs0, tid);            // prologue: stage 0

    for (int s = 0; s < nst; ++s) {
        __syncthreads();                        // all waves done with stage s-1
        const int cur = s & 1;
        if (s + 1 < nst) {
            gemm_stage(A, Bt, bm, bn, K, (s + 1) * GBK,
                       as0 + (uint32_t)((1 - cur) * ASZ) * 2,
                       bs0 + (uint32_t)((1 - cur) * BSZ) * 2, tid);
            wait_async<6>();                    // stage s copies complete
        } else {
            wait_async<0>();
        }
        __syncthreads();                        // stage s visible to all waves

        const bf16_t* Ab = As + cur * ASZ;
        const bf16_t* Bb = Bs + cur * BSZ;
#pragma unroll
        for (int ks = 0; ks < 2; ++ks) {
            v16bf af[2], bfb[2];
#pragma unroll
            for (int im = 0; im < 2; ++im)
                af[im] = load_frag_a(&Ab[(wm * 32 + im * 16) * ASTR + ks * 32], ASTR, lane);
#pragma unroll
            for (int in = 0; in < 2; ++in)
                bfb[in] = load_frag_b(&Bb[(wn * 32 + in * 16) * BSTR + ks * 32], BSTR, lane);
#pragma unroll
            for (int im = 0; im < 2; ++im)
#pragma unroll
                for (int in = 0; in < 2; ++in)
                    acc[im][in] = WMMA_BF16(af[im], bfb[in], acc[im][in]);
        }
    }

    const int half = lane >> 4;
    const int cn   = lane & 15;
#pragma unroll
    for (int im = 0; im < 2; ++im) {
#pragma unroll
        for (int in = 0; in < 2; ++in) {
            const int gn = bn + wn * 32 + in * 16 + cn;
            const float bv = bias[gn];
#pragma unroll
            for (int i = 0; i < 8; ++i) {
                const int gm = bm + wm * 32 + im * 16 + i + 8 * half;
                const float v = acc[im][in][i] + bv;
                if (OUT_BF16)
                    ((bf16_t*)Cout)[(size_t)gm * N + gn] = (bf16_t)v;
                else
                    ((float*)Cout)[(size_t)gm * N + gn] = v;
            }
        }
    }
}

// ---------------------------------------------------------------------------
// Flash attention, causal. Block = (b, h, 64-query tile); 128 threads, 4 waves.
// Double-buffered K (async) and V (load + LDS transpose) prefetch.
// ---------------------------------------------------------------------------
#define HD   64
#define KSTR (HD + 8)           // 72
#define TSZ  (64 * KSTR)

__device__ __forceinline__ void attn_stage_kv(const bf16_t* qkv, size_t rowbase,
                                              size_t rs, int E, int kj,
                                              uint32_t ks_lds, bf16_t* Vsb, int tid) {
    const bf16_t* kb = qkv + (rowbase + (size_t)kj * 64) * rs + E;
    const bf16_t* vb = kb + E;
    // K tile: async copy, natural [key][d]: 512 x 16B chunks, 4 per thread
#pragma unroll
    for (int i = 0; i < 4; ++i) {
        const int c = tid + i * 128;
        const int r = c >> 3, seg = c & 7;
        async_copy_b128(ks_lds + (uint32_t)(r * KSTR + seg * 8) * 2,
                        kb + (size_t)r * rs + seg * 8);
    }
    // V tile: transpose into [d][key]
#pragma unroll
    for (int i = 0; i < 4; ++i) {
        const int c = tid + i * 128;
        const int r = c >> 3, seg = c & 7;
        const v8bf v = *(const v8bf*)(vb + (size_t)r * rs + seg * 8);
#pragma unroll
        for (int e = 0; e < 8; ++e)
            Vsb[(seg * 8 + e) * KSTR + r] = v[e];
    }
}

__global__ __launch_bounds__(128) void flash_attn_wmma(const bf16_t* __restrict__ qkv,
                                                       bf16_t* __restrict__ y,
                                                       int T, int E) {
    __shared__ __align__(16) bf16_t Qs[TSZ];
    __shared__ __align__(16) bf16_t Ks[2 * TSZ];     // [key][d] x2
    __shared__ __align__(16) bf16_t Vs[2 * TSZ];     // [d][key] x2
    __shared__ __align__(16) bf16_t Ps[4 * 16 * KSTR];

    const int tid  = threadIdx.x;
    const int lane = tid & 31;
    const int wave = tid >> 5;
    const int qt   = blockIdx.x;
    const int bh   = blockIdx.y;
    const int b    = bh >> 4;
    const int h    = bh & 15;
    const int q0   = qt * 64;
    const size_t rs = 3 * (size_t)E;
    const size_t rowbase = (size_t)b * T;     // +t gives qkv row
    const size_t hoff = (size_t)h * HD;

    const bf16_t* qb = qkv + (rowbase + q0) * rs + hoff;
    const uint32_t qs0 = lds_addr32(Qs);
    const uint32_t ks0 = lds_addr32(Ks);

    // Q tile async
#pragma unroll
    for (int i = 0; i < 4; ++i) {
        const int c = tid + i * 128;
        const int r = c >> 3, seg = c & 7;
        async_copy_b128(qs0 + (uint32_t)(r * KSTR + seg * 8) * 2,
                        qb + (size_t)r * rs + seg * 8);
    }

    const int nkt = qt + 1;
    attn_stage_kv(qkv + hoff, rowbase, rs, E, 0, ks0, Vs, tid);  // stage 0

    float m_run[8], l_run[8];
    v8f o[4] = {};
#pragma unroll
    for (int i = 0; i < 8; ++i) { m_run[i] = -3.0e38f; l_run[i] = 0.0f; }

    const int half = lane >> 4;
    const int cn   = lane & 15;

    for (int kj = 0; kj < nkt; ++kj) {
        __syncthreads();                          // all waves done with stage kj-1
        const int cur = kj & 1;
        if (kj + 1 < nkt) {
            attn_stage_kv(qkv + hoff, rowbase, rs, E, kj + 1,
                          ks0 + (uint32_t)((1 - cur) * TSZ) * 2,
                          Vs + (1 - cur) * TSZ, tid);
            wait_async<4>();                      // stage kj (and Q) complete
        } else {
            wait_async<0>();
        }
        __syncthreads();

        const bf16_t* Kb = Ks + cur * TSZ;
        const bf16_t* Vb = Vs + cur * TSZ;

        // S = (Q Kt) * 1/8
        v8f s[4] = {};
        const v16bf aq0 = load_frag_a(&Qs[(wave * 16) * KSTR], KSTR, lane);
        const v16bf aq1 = load_frag_a(&Qs[(wave * 16) * KSTR + 32], KSTR, lane);
#pragma unroll
        for (int t = 0; t < 4; ++t) {
            const v16bf bk0 = load_frag_b(&Kb[(t * 16) * KSTR], KSTR, lane);
            const v16bf bk1 = load_frag_b(&Kb[(t * 16) * KSTR + 32], KSTR, lane);
            s[t] = WMMA_BF16(aq0, bk0, s[t]);
            s[t] = WMMA_BF16(aq1, bk1, s[t]);
#pragma unroll
            for (int i = 0; i < 8; ++i) s[t][i] *= 0.125f;
        }

        if (kj == qt) {                           // causal mask on diagonal tile
#pragma unroll
            for (int t = 0; t < 4; ++t) {
                const int col = t * 16 + cn;
#pragma unroll
                for (int i = 0; i < 8; ++i) {
                    const int row = wave * 16 + i + 8 * half;
                    if (col > row) s[t][i] = -3.0e38f;
                }
            }
        }

        // Online softmax
        float mt[8];
#pragma unroll
        for (int i = 0; i < 8; ++i) {
            float v = fmaxf(fmaxf(s[0][i], s[1][i]), fmaxf(s[2][i], s[3][i]));
#pragma unroll
            for (int m = 8; m >= 1; m >>= 1) v = fmaxf(v, __shfl_xor(v, m, 32));
            mt[i] = v;
        }
        float alpha[8];
#pragma unroll
        for (int i = 0; i < 8; ++i) {
            const float mn = fmaxf(m_run[i], mt[i]);
            alpha[i] = __expf(m_run[i] - mn);
            m_run[i] = mn;
        }
        float rsum[8];
#pragma unroll
        for (int i = 0; i < 8; ++i) {
            float acc = 0.0f;
#pragma unroll
            for (int t = 0; t < 4; ++t) {
                const float p = __expf(s[t][i] - m_run[i]);
                s[t][i] = p;
                acc += p;
            }
#pragma unroll
            for (int m = 8; m >= 1; m >>= 1) acc += __shfl_xor(acc, m, 32);
            rsum[i] = acc;
        }
#pragma unroll
        for (int i = 0; i < 8; ++i) l_run[i] = l_run[i] * alpha[i] + rsum[i];
#pragma unroll
        for (int c = 0; c < 4; ++c)
#pragma unroll
            for (int i = 0; i < 8; ++i) o[c][i] *= alpha[i];

        // C-layout -> A-layout via wave-private LDS strip
        bf16_t* pw = &Ps[wave * 16 * KSTR];
#pragma unroll
        for (int t = 0; t < 4; ++t)
#pragma unroll
            for (int i = 0; i < 8; ++i)
                pw[(i + 8 * half) * KSTR + t * 16 + cn] = (bf16_t)s[t][i];

        const v16bf ap0 = load_frag_a(pw, KSTR, lane);
        const v16bf ap1 = load_frag_a(pw + 32, KSTR, lane);
#pragma unroll
        for (int c = 0; c < 4; ++c) {
            const v16bf bv0 = load_frag_b(&Vb[(c * 16) * KSTR], KSTR, lane);
            const v16bf bv1 = load_frag_b(&Vb[(c * 16) * KSTR + 32], KSTR, lane);
            o[c] = WMMA_BF16(ap0, bv0, o[c]);
            o[c] = WMMA_BF16(ap1, bv1, o[c]);
        }
    }

    float invl[8];
#pragma unroll
    for (int i = 0; i < 8; ++i) invl[i] = 1.0f / l_run[i];
#pragma unroll
    for (int c = 0; c < 4; ++c)
#pragma unroll
        for (int i = 0; i < 8; ++i) {
            const int row = q0 + wave * 16 + i + 8 * half;
            const int col = h * HD + c * 16 + cn;
            y[((size_t)b * T + row) * E + col] = (bf16_t)(o[c][i] * invl[i]);
        }
}

// ---------------------------------------------------------------------------
// Launch: converts -> qkv GEMM -> flash attention -> proj GEMM
// ---------------------------------------------------------------------------
extern "C" void kernel_launch(void* const* d_in, const int* in_sizes, int n_in,
                              void* d_out, int out_size, void* d_ws, size_t ws_size,
                              hipStream_t stream) {
    (void)in_sizes; (void)n_in; (void)out_size; (void)ws_size;
    const float* x      = (const float*)d_in[0];
    const float* W_attn = (const float*)d_in[1];
    const float* b_attn = (const float*)d_in[2];
    const float* W_proj = (const float*)d_in[3];
    const float* b_proj = (const float*)d_in[4];
    float*       out    = (float*)d_out;

    const int Bb = 2, T = 2048, E = 1024;
    const int M = Bb * T;                       // 4096

    char* ws = (char*)d_ws;
    bf16_t* qkv  = (bf16_t*)(ws);                                   // M x 3E   24 MB
    bf16_t* ybuf = (bf16_t*)(ws + (size_t)M * 3 * E * 2);           // M x E     8 MB
    bf16_t* xb   = (bf16_t*)(ws + (size_t)M * 3 * E * 2 + (size_t)M * E * 2);      // 8 MB
    bf16_t* WaT  = (bf16_t*)((char*)xb + (size_t)M * E * 2);        // 3E x E    6 MB
    bf16_t* WpT  = (bf16_t*)((char*)WaT + (size_t)3 * E * E * 2);   // E x E     2 MB

    cvt_bf16<<<dim3((M * E / 4 + 255) / 256), 256, 0, stream>>>(x, xb, M * E / 4);
    transpose_cvt<<<dim3(3 * E / 32, E / 32), dim3(32, 8), 0, stream>>>(W_attn, WaT, E, 3 * E);
    transpose_cvt<<<dim3(E / 32, E / 32), dim3(32, 8), 0, stream>>>(W_proj, WpT, E, E);

    gemm_bias_wmma<true><<<dim3(3 * E / GBN, M / GBM), 256, 0, stream>>>(
        xb, WaT, b_attn, qkv, M, 3 * E, E);

    flash_attn_wmma<<<dim3(T / 64, Bb * 16), 128, 0, stream>>>(qkv, ybuf, T, E);

    gemm_bias_wmma<false><<<dim3(E / GBN, M / GBM), 256, 0, stream>>>(
        ybuf, WpT, b_proj, out, M, E, E);
}
